// GRULayer_30159260353161
// MI455X (gfx1250) — matrix operand order repaired and checked by reading, not compiled, not run on hardware
//
#include <hip/hip_runtime.h>
#include <hip/hip_bf16.h>
#include <math.h>

// ---------------------------------------------------------------------------
// GRU layer for MI455X (gfx1250): bf16 WMMA GEMMs + fused LayerNorm epilogue,
// persistent single-kernel recurrence with device-scope grid barrier.
// B=32, T=1000, I=512, H=1024.
// ---------------------------------------------------------------------------

#define GRU_B 32
#define GRU_T 1000
#define GRU_I 512
#define GRU_H 1024
#define LN_EPS 1e-5f
#define REC_BLOCKS 64          // persistent recurrence: 64 blocks x 4 waves = 256 waves

typedef __attribute__((ext_vector_type(16))) __bf16 bf16x16;
typedef __attribute__((ext_vector_type(8)))  __bf16 bf16x8;
typedef __attribute__((ext_vector_type(8)))  float  v8f;

// float -> bf16, round-to-nearest-even
__device__ __forceinline__ __bf16 f2bf(float f) {
    unsigned u = __builtin_bit_cast(unsigned, f);
    u += 0x7FFFu + ((u >> 16) & 1u);
    unsigned short h = (unsigned short)(u >> 16);
    return __builtin_bit_cast(__bf16, h);
}

// A-fragment (16x32 bf16) per ISA 7.12.2:
// lanes 0-15 : row = lane,    halves 0-7 -> K 0..7,  halves 8-15 -> K 16..23
// lanes16-31 : row = lane-16, halves 0-7 -> K 8..15, halves 8-15 -> K 24..31
__device__ __forceinline__ bf16x16 load_a_frag(const __bf16* A, int lda,
                                               int row_base, int k_base, int lane) {
    int r    = row_base + (lane & 15);
    int koff = (lane >> 4) * 8;
    const __bf16* p = A + (size_t)r * lda + (size_t)(k_base + koff);
    bf16x8 lo = *(const bf16x8*)p;
    bf16x8 hi = *(const bf16x8*)(p + 16);
    bf16x16 a;
#pragma unroll
    for (int i = 0; i < 8; ++i) { a[i] = lo[i]; a[i + 8] = hi[i]; }
    return a;
}

// B-fragment (32x16 bf16), B[k][n] = Wrow[n][k] (weights stored [N,K] row-major)
__device__ __forceinline__ bf16x16 load_b_frag(const __bf16* Wt, int ldb,
                                               int n_base, int k_base, int lane) {
    int n    = n_base + (lane & 15);
    int koff = (lane >> 4) * 8;
    const __bf16* p = Wt + (size_t)n * ldb + (size_t)(k_base + koff);
    bf16x8 lo = *(const bf16x8*)p;
    bf16x8 hi = *(const bf16x8*)(p + 16);
    bf16x16 b;
#pragma unroll
    for (int i = 0; i < 8; ++i) { b[i] = lo[i]; b[i + 8] = hi[i]; }
    return b;
}

__device__ __forceinline__ v8f wmma_bf16(bf16x16 a, bf16x16 b, v8f c) {
    return __builtin_amdgcn_wmma_f32_16x16x32_bf16(
        false, a, false, b, (short)0, c, false, false);
}

// Single 16x16 tile GEMM over K (compile-time, multiple of 64).
// Ping-pong double buffering: two fragment sets alternate compute/load so the
// register allocator never has to copy fragments (no v_mov rotation).
template <int K>
__device__ __forceinline__ v8f gemm_tile(const __bf16* __restrict__ A,
                                         const __bf16* __restrict__ Bm,
                                         int m_base, int n_base, int lane) {
    static_assert(K % 64 == 0 && K >= 64, "K must be a multiple of 64");
    v8f c = (v8f){0.f,0.f,0.f,0.f,0.f,0.f,0.f,0.f};
    bf16x16 aA = load_a_frag(A, K, m_base, 0,  lane);
    bf16x16 bA = load_b_frag(Bm, K, n_base, 0,  lane);
    bf16x16 aB = load_a_frag(A, K, m_base, 32, lane);
    bf16x16 bB = load_b_frag(Bm, K, n_base, 32, lane);
    for (int k = 64; k < K; k += 64) {
        c  = wmma_bf16(aA, bA, c);
        aA = load_a_frag(A, K, m_base, k, lane);
        bA = load_b_frag(Bm, K, n_base, k, lane);
        c  = wmma_bf16(aB, bB, c);
        aB = load_a_frag(A, K, m_base, k + 32, lane);
        bB = load_b_frag(Bm, K, n_base, k + 32, lane);
    }
    c = wmma_bf16(aA, bA, c);
    return wmma_bf16(aB, bB, c);
}

// ---------------------------------------------------------------------------
// f32 -> bf16 conversion
// ---------------------------------------------------------------------------
__global__ void __launch_bounds__(256) cvt_bf16_kernel(const float* __restrict__ src,
                                                       __bf16* __restrict__ dst, int n) {
    int i = blockIdx.x * 256 + threadIdx.x;
    if (i < n) dst[i] = f2bf(src[i]);
}

// ---------------------------------------------------------------------------
// Phase 1: preact = LayerNorm(x @ W^T) * g + b  for W / Wz / Wr.
// grid = (2000, 3), block = 512 (16 waves). Block owns 16 rows x full H.
// Wave w: 16x64 strip (4 WMMA accumulators); K loop ping-pong pipelined.
// ---------------------------------------------------------------------------
__global__ void __launch_bounds__(512)
gemm_ln_kernel(const __bf16* __restrict__ xb,
               const __bf16* __restrict__ Wb,  const __bf16* __restrict__ Wzb,
               const __bf16* __restrict__ Wrb,
               const float* __restrict__ g,  const float* __restrict__ bb,
               const float* __restrict__ gz, const float* __restrict__ bz,
               const float* __restrict__ gr, const float* __restrict__ br,
               float* __restrict__ wx, float* __restrict__ wzx,
               float* __restrict__ wrx) {
    const __bf16* Wsel; const float* gsel; const float* bsel; float* osel;
    if (blockIdx.y == 0)      { Wsel = Wb;  gsel = g;  bsel = bb; osel = wx;  }
    else if (blockIdx.y == 1) { Wsel = Wzb; gsel = gz; bsel = bz; osel = wzx; }
    else                      { Wsel = Wrb; gsel = gr; bsel = br; osel = wrx; }

    const int tid  = threadIdx.x;
    const int wave = tid >> 5;
    const int lane = tid & 31;
    const int row_base = blockIdx.x * 16;
    const int n_base   = wave * 64;

    v8f c[4];
#pragma unroll
    for (int j = 0; j < 4; ++j) c[j] = (v8f){0.f,0.f,0.f,0.f,0.f,0.f,0.f,0.f};

    // ping-pong pipelined K loop: (A + 4 B) fragment sets alternate
    bf16x16 aA = load_a_frag(xb, GRU_I, row_base, 0, lane);
    bf16x16 bA[4];
#pragma unroll
    for (int j = 0; j < 4; ++j) bA[j] = load_b_frag(Wsel, GRU_I, n_base + 16 * j, 0, lane);
    bf16x16 aB = load_a_frag(xb, GRU_I, row_base, 32, lane);
    bf16x16 bB[4];
#pragma unroll
    for (int j = 0; j < 4; ++j) bB[j] = load_b_frag(Wsel, GRU_I, n_base + 16 * j, 32, lane);

    for (int k = 64; k < GRU_I; k += 64) {
#pragma unroll
        for (int j = 0; j < 4; ++j) c[j] = wmma_bf16(aA, bA[j], c[j]);
        aA = load_a_frag(xb, GRU_I, row_base, k, lane);
#pragma unroll
        for (int j = 0; j < 4; ++j) bA[j] = load_b_frag(Wsel, GRU_I, n_base + 16 * j, k, lane);
#pragma unroll
        for (int j = 0; j < 4; ++j) c[j] = wmma_bf16(aB, bB[j], c[j]);
        aB = load_a_frag(xb, GRU_I, row_base, k + 32, lane);
#pragma unroll
        for (int j = 0; j < 4; ++j) bB[j] = load_b_frag(Wsel, GRU_I, n_base + 16 * j, k + 32, lane);
    }
#pragma unroll
    for (int j = 0; j < 4; ++j) c[j] = wmma_bf16(aA, bA[j], c[j]);
#pragma unroll
    for (int j = 0; j < 4; ++j) c[j] = wmma_bf16(aB, bB[j], c[j]);

    __shared__ float tile[16][GRU_H];   // 64 KB
    const int hi = lane >> 4;
    const int nl = lane & 15;
#pragma unroll
    for (int j = 0; j < 4; ++j)
#pragma unroll
        for (int v = 0; v < 8; ++v)
            tile[v + 8 * hi][n_base + 16 * j + nl] = c[j][v];
    __syncthreads();

    // LayerNorm: wave w handles row w
    const int row = wave;
    float s = 0.f, s2 = 0.f;
    for (int i = lane; i < GRU_H; i += 32) {
        float v = tile[row][i];
        s += v; s2 += v * v;
    }
#pragma unroll
    for (int off = 16; off > 0; off >>= 1) {
        s  += __shfl_xor(s,  off, 32);
        s2 += __shfl_xor(s2, off, 32);
    }
    const float mean = s * (1.0f / GRU_H);
    const float var  = s2 * (1.0f / GRU_H) - mean * mean;
    const float inv  = rsqrtf(var + LN_EPS);

    const size_t orow = (size_t)(row_base + row) * GRU_H;
    for (int i = lane; i < GRU_H; i += 32)
        osel[orow + i] = (tile[row][i] - mean) * inv * gsel[i] + bsel[i];
}

// ---------------------------------------------------------------------------
// state + barrier init (re-run every launch -> deterministic graph replays)
// ---------------------------------------------------------------------------
__global__ void __launch_bounds__(256) init_kernel(float* __restrict__ yf,
                                                   __bf16* __restrict__ ybf,
                                                   unsigned* __restrict__ bar) {
    int i = blockIdx.x * 256 + threadIdx.x;
    if (i < GRU_B * GRU_H) { yf[i] = 0.f; ybf[i] = f2bf(0.f); }
    if (i == 0) *bar = 0u;
}

// ---------------------------------------------------------------------------
// Device-scope grid barrier: release add + acquire spin (s_sleep while waiting).
// ---------------------------------------------------------------------------
__device__ __forceinline__ void grid_sync(unsigned* bar, int idx) {
    __syncthreads();
    if (threadIdx.x == 0) {
        __hip_atomic_fetch_add(bar, 1u, __ATOMIC_RELEASE, __HIP_MEMORY_SCOPE_AGENT);
        const unsigned target = (unsigned)(idx + 1) * REC_BLOCKS;
        while (__hip_atomic_load(bar, __ATOMIC_ACQUIRE, __HIP_MEMORY_SCOPE_AGENT) < target)
            __builtin_amdgcn_s_sleep(2);
    }
    __syncthreads();
}

// ---------------------------------------------------------------------------
// Persistent recurrence: one kernel for all T steps.
// 64 blocks x 128 threads = 256 waves (co-resident).
// Phase A (256 tiles): z = sig(wzx + y Vz^T), r = sig(wrx + y Vr^T); a = bf16(r*y)
// Phase B (128 tiles): c = tanh(wx + a V^T); y = z*y + (1-z)*c
// ---------------------------------------------------------------------------
__global__ void __launch_bounds__(128)
recurrence_kernel(__bf16* __restrict__ ybf, float* __restrict__ yf,
                  const __bf16* __restrict__ Vb,  const __bf16* __restrict__ Vzb,
                  const __bf16* __restrict__ Vrb,
                  const float* __restrict__ wx,  const float* __restrict__ wzx,
                  const float* __restrict__ wrx,
                  float* __restrict__ zbuf, __bf16* __restrict__ abf,
                  float* __restrict__ out, unsigned* __restrict__ bar) {
    const int lane = threadIdx.x & 31;
    const int gw   = blockIdx.x * 4 + (threadIdx.x >> 5);   // 0..255
    const int isr  = gw >> 7;                               // phase A: 0 = z, 1 = r
    const int remA = gw & 127;
    const int mtA  = remA >> 6;
    const int ntA  = remA & 63;
    const int mtB  = gw >> 6;                               // phase B (gw < 128)
    const int ntB  = gw & 63;
    const int hi   = lane >> 4;
    const int nl   = lane & 15;

    const __bf16* VselA = isr ? Vrb : Vzb;
    const float*  preA  = isr ? wrx : wzx;

    int bidx = 0;
    for (int t = 0; t < GRU_T; ++t) {
        // ---- phase A: gates z and r ----
        {
            v8f c = gemm_tile<GRU_H>(ybf, VselA, mtA * 16, ntA * 16, lane);
            const int n = ntA * 16 + nl;
#pragma unroll
            for (int v = 0; v < 8; ++v) {
                int m = mtA * 16 + v + 8 * hi;
                float pa = preA[((size_t)m * GRU_T + t) * GRU_H + n];
                float sg = 1.0f / (1.0f + __expf(-(pa + c[v])));
                size_t idx = (size_t)m * GRU_H + n;
                if (isr) abf[idx]  = f2bf(sg * yf[idx]);
                else     zbuf[idx] = sg;
            }
        }
        grid_sync(bar, bidx++);

        // ---- phase B: candidate + state update ----
        if (gw < 128) {
            v8f c = gemm_tile<GRU_H>(abf, Vb, mtB * 16, ntB * 16, lane);
            const int n = ntB * 16 + nl;
#pragma unroll
            for (int v = 0; v < 8; ++v) {
                int m = mtB * 16 + v + 8 * hi;
                size_t idx  = (size_t)m * GRU_H + n;
                size_t oidx = ((size_t)m * GRU_T + t) * GRU_H + n;
                float ct = tanhf(wx[oidx] + c[v]);
                float z  = zbuf[idx];
                float yn = z * yf[idx] + (1.0f - z) * ct;
                yf[idx]   = yn;
                ybf[idx]  = f2bf(yn);
                out[oidx] = yn;
            }
        }
        grid_sync(bar, bidx++);
    }
}

// ---------------------------------------------------------------------------
// Host-side orchestration
// ---------------------------------------------------------------------------
static inline size_t align_up(size_t x, size_t a) { return (x + a - 1) & ~(a - 1); }

extern "C" void kernel_launch(void* const* d_in, const int* in_sizes, int n_in,
                              void* d_out, int out_size, void* d_ws, size_t ws_size,
                              hipStream_t stream) {
    (void)in_sizes; (void)n_in; (void)out_size; (void)ws_size;
    const float* x  = (const float*)d_in[0];
    const float* W  = (const float*)d_in[1];
    const float* Wz = (const float*)d_in[2];
    const float* Wr = (const float*)d_in[3];
    const float* V  = (const float*)d_in[4];
    const float* Vz = (const float*)d_in[5];
    const float* Vr = (const float*)d_in[6];
    const float* g  = (const float*)d_in[7];
    const float* b  = (const float*)d_in[8];
    const float* gz = (const float*)d_in[9];
    const float* bz = (const float*)d_in[10];
    const float* gr = (const float*)d_in[11];
    const float* br = (const float*)d_in[12];
    float* out = (float*)d_out;

    const size_t nx  = (size_t)GRU_B * GRU_T * GRU_I;
    const size_t nw  = (size_t)GRU_H * GRU_I;
    const size_t nv  = (size_t)GRU_H * GRU_H;
    const size_t npr = (size_t)GRU_B * GRU_T * GRU_H;
    const size_t ny  = (size_t)GRU_B * GRU_H;

    char* p = (char*)d_ws; size_t off = 0;
    auto carve = [&](size_t bytes) -> char* {
        off = align_up(off, 256); char* r = p + off; off += bytes; return r;
    };
    __bf16*   xb   = (__bf16*)  carve(nx  * 2);
    __bf16*   Wb   = (__bf16*)  carve(nw  * 2);
    __bf16*   Wzb  = (__bf16*)  carve(nw  * 2);
    __bf16*   Wrb  = (__bf16*)  carve(nw  * 2);
    __bf16*   Vb   = (__bf16*)  carve(nv  * 2);
    __bf16*   Vzb  = (__bf16*)  carve(nv  * 2);
    __bf16*   Vrb  = (__bf16*)  carve(nv  * 2);
    float*    wx   = (float*)   carve(npr * 4);
    float*    wzx  = (float*)   carve(npr * 4);
    float*    wrx  = (float*)   carve(npr * 4);
    float*    yf   = (float*)   carve(ny  * 4);
    __bf16*   ybf  = (__bf16*)  carve(ny  * 2);
    float*    zbuf = (float*)   carve(ny  * 4);
    __bf16*   abf  = (__bf16*)  carve(ny  * 2);
    unsigned* bar  = (unsigned*)carve(256);

    // 1) convert operands to bf16
    cvt_bf16_kernel<<<(nx + 255) / 256, 256, 0, stream>>>(x,  xb,  (int)nx);
    cvt_bf16_kernel<<<(nw + 255) / 256, 256, 0, stream>>>(W,  Wb,  (int)nw);
    cvt_bf16_kernel<<<(nw + 255) / 256, 256, 0, stream>>>(Wz, Wzb, (int)nw);
    cvt_bf16_kernel<<<(nw + 255) / 256, 256, 0, stream>>>(Wr, Wrb, (int)nw);
    cvt_bf16_kernel<<<(nv + 255) / 256, 256, 0, stream>>>(V,  Vb,  (int)nv);
    cvt_bf16_kernel<<<(nv + 255) / 256, 256, 0, stream>>>(Vz, Vzb, (int)nv);
    cvt_bf16_kernel<<<(nv + 255) / 256, 256, 0, stream>>>(Vr, Vrb, (int)nv);

    // 2) fused GEMM + LayerNorm for all three projections
    dim3 gdim((GRU_B * GRU_T) / 16, 3);
    gemm_ln_kernel<<<gdim, 512, 0, stream>>>(xb, Wb, Wzb, Wrb,
                                             g, b, gz, bz, gr, br,
                                             wx, wzx, wrx);

    // 3) persistent recurrence (single kernel, grid barrier between phases)
    init_kernel<<<(ny + 255) / 256, 256, 0, stream>>>(yf, ybf, bar);
    recurrence_kernel<<<REC_BLOCKS, 128, 0, stream>>>(ybf, yf, Vb, Vzb, Vrb,
                                                      wx, wzx, wrx,
                                                      zbuf, abf, out, bar);
}
